// SelfAttention_Inefficient_34694745817385
// MI455X (gfx1250) — matrix operand order repaired and checked by reading, not compiled, not run on hardware
//
#include <hip/hip_runtime.h>

// Problem constants (match reference)
#define BB 4
#define NN 4096
#define DD 64
#define HH 2

typedef __attribute__((ext_vector_type(16))) __bf16 v16bf;
typedef __attribute__((ext_vector_type(8)))  __bf16 v8bf;
typedef __attribute__((ext_vector_type(8)))  float  v8f;

__device__ __forceinline__ v8f wmma_bf16(v16bf a, v16bf b, v8f c) {
  // 8 args: (neg_a, A, neg_b, B, c_mod, C, reuse_a, reuse_b)
  return __builtin_amdgcn_wmma_f32_16x16x32_bf16(false, a, false, b, (short)0, c,
                                                 false, false);
}

// ---------------------------------------------------------------------------
// Kernel 1: Q/K/V projections.
//   ws layout: Q' [B,H,N,D] bf16 | K [B,H,N,D] bf16 | V^T [B,H,D,N] bf16
//   where Q' = Q * (1/N). 1/N = 2^-12 is a power of two: scaling is exact
//   and commutes with relu, so relu(Q'K^T) == relu(QK^T)/N bit-exactly.
// Grid: B*(N/64) blocks of 128 threads (4 wave32); each wave does 16 rows.
// All six weight matrices staged once per block into LDS as bf16,
// pre-transposed so B-fragments are contiguous 32B LDS vector reads.
// All matrix loops fully unrolled -> straight-line 48-WMMA body, no branches.
// ---------------------------------------------------------------------------
__global__ __launch_bounds__(128) void qkv_project_kernel(
    const float* __restrict__ x,
    const float* __restrict__ WQ,
    const float* __restrict__ WK,
    const float* __restrict__ WV,
    __bf16* __restrict__ ws)
{
  __shared__ __bf16 wlds[6][DD * DD];   // 48 KB: wlds[m][n*D + k] = W_m[k][n]
  // m order: 0=WQ.h0 1=WQ.h1 2=WK.h0 3=WK.h1 4=WV.h0 5=WV.h1

  const int lane = threadIdx.x & 31;
  const int wave = threadIdx.x >> 5;
  const int tilesPerB = NN / 64;
  const int b = blockIdx.x / tilesPerB;
  const int rowBase = (blockIdx.x % tilesPerB) * 64 + wave * 16;
  const int l15 = lane & 15;
  const int lhi = lane >> 4;

  // Cooperative stage: coalesced f32 reads, transpose+convert into LDS.
  const float* Wsrc[6] = { WQ, WQ + DD * DD, WK, WK + DD * DD, WV, WV + DD * DD };
  #pragma unroll
  for (int m = 0; m < 6; ++m) {
    const float* W = Wsrc[m];
    for (int idx = threadIdx.x; idx < DD * DD; idx += 128) {
      const int k = idx >> 6;   // contraction dim (row of W)
      const int n = idx & 63;   // output column
      wlds[m][n * DD + k] = (__bf16)W[idx];
    }
  }
  __syncthreads();

  // A-fragments of x: 32B vector loads (A-layout K runs are contiguous).
  const float* xrow = x + (((size_t)b * NN + rowBase) + l15) * DD + lhi * 16;
  const v8f x0 = *(const v8f*)(xrow);
  const v8f x1 = *(const v8f*)(xrow + 8);
  const v8f x2 = *(const v8f*)(xrow + 32);
  const v8f x3 = *(const v8f*)(xrow + 40);
  v16bf a0, a1;
  #pragma unroll
  for (int e = 0; e < 8; ++e) {
    a0[e] = (__bf16)x0[e]; a0[8 + e] = (__bf16)x1[e];   // d = 0..31
    a1[e] = (__bf16)x2[e]; a1[8 + e] = (__bf16)x3[e];   // d = 32..63
  }

  const size_t qkvStride = (size_t)BB * HH * NN * DD;
  const int kb = lhi * 16;     // contiguous-order B K base (k = 16*lhi + s)
  const int rb = 8 * lhi;      // ISA C layout M base
  const float invN = 1.0f / (float)NN;   // exact power of two

  // ---- Q' and K (row-major outputs). m = mt*2 + h for mt in {0,1}.
  #pragma unroll
  for (int m = 0; m < 4; ++m) {
    const int mt = m >> 1;                       // 0 = Q', 1 = K
    const int h  = m & 1;
    const float oscale = (m < 2) ? invN : 1.0f;  // compile-time after unroll
    __bf16* O = ws + (size_t)mt * qkvStride +
                (((size_t)b * HH + h) * NN + rowBase) * DD;
    #pragma unroll
    for (int ct = 0; ct < 4; ++ct) {
      const int col = ct * 16 + l15;
      const v16bf b0 = *(const v16bf*)&wlds[m][(size_t)col * DD + kb];
      const v16bf b1 = *(const v16bf*)&wlds[m][(size_t)col * DD + 32 + kb];
      v8f c = {};
      c = wmma_bf16(a0, b0, c);
      c = wmma_bf16(a1, b1, c);
      #pragma unroll
      for (int r = 0; r < 8; ++r)
        O[(size_t)(rb + r) * DD + col] = (__bf16)(c[r] * oscale);
    }
  }

  // ---- V, stored transposed [B,H,D,N]; 8 rows/lane are contiguous.
  #pragma unroll
  for (int h = 0; h < HH; ++h) {
    const int m = 4 + h;
    #pragma unroll
    for (int ct = 0; ct < 4; ++ct) {
      const int col = ct * 16 + l15;
      const v16bf b0 = *(const v16bf*)&wlds[m][(size_t)col * DD + kb];
      const v16bf b1 = *(const v16bf*)&wlds[m][(size_t)col * DD + 32 + kb];
      v8f c = {};
      c = wmma_bf16(a0, b0, c);
      c = wmma_bf16(a1, b1, c);
      __bf16* OT = ws + 2 * qkvStride +
                   (((size_t)b * HH + h) * DD + col) * NN + rowBase;
      v8bf pk;
      #pragma unroll
      for (int r = 0; r < 8; ++r) pk[r] = (__bf16)c[r];
      *(v8bf*)(OT + rb) = pk;
    }
  }
}

// ---------------------------------------------------------------------------
// Kernel 2: streaming relu-attention + residual.
// Grid: B*(N/128) blocks of 128 threads; each wave owns TWO 16-row i-tiles
// (32 rows) and 2x4 persistent f32 C-fragments across all j and both heads.
// K^T / V^T fragments are i-independent: 8 x 32B loads feed 16 WMMAs/iter.
// S staged per-wave in LDS as bf16 (relu pre-store; 1/N folded into Q'),
// so the C->A transpose costs 16 b16 stores + 2 b128 loads per tile with no
// block barrier (per-wave buffer + in-order LDS pipe). Prefetch unguarded:
// last-iteration addresses stay inside ws, and ISA prefetch is speculative.
// ---------------------------------------------------------------------------
__global__ __launch_bounds__(128) void relu_attn_kernel(
    const float* __restrict__ x,
    const __bf16* __restrict__ ws,
    float* __restrict__ out)
{
  __shared__ __bf16 sstage[4][2][16 * 32];   // [wave][itile][16x32 S] = 8 KB

  const int lane = threadIdx.x & 31;
  const int wave = threadIdx.x >> 5;
  const int tilesPerB = NN / 128;
  const int b = blockIdx.x / tilesPerB;
  const int rowBase = (blockIdx.x % tilesPerB) * 128 + wave * 32;

  const int l15 = lane & 15;
  const int lhi = lane >> 4;

  const size_t qkvStride = (size_t)BB * HH * NN * DD;
  const __bf16* Qbase  = ws;
  const __bf16* Kbase  = ws + qkvStride;
  const __bf16* VTbase = ws + 2 * qkvStride;

  v8f acc[2][4] = {};   // two 16x64 f32 output tiles, summed over heads

  for (int h = 0; h < HH; ++h) {
    const __bf16* Qh  = Qbase  + (((size_t)b * HH + h) * NN + rowBase) * DD;
    const __bf16* Kh  = Kbase  + (((size_t)b * HH + h) * NN) * DD;
    const __bf16* VTh = VTbase + (((size_t)b * HH + h) * DD) * NN;

    // Q' A-fragments for both i-tiles (vector loads, resident for the j loop)
    v16bf qa[2][2];
    #pragma unroll
    for (int t = 0; t < 2; ++t) {
      const __bf16* qrow = Qh + (size_t)(t * 16 + l15) * DD + lhi * 16;
      const v8bf q0 = *(const v8bf*)(qrow);
      const v8bf q1 = *(const v8bf*)(qrow + 8);
      const v8bf q2 = *(const v8bf*)(qrow + 32);
      const v8bf q3 = *(const v8bf*)(qrow + 40);
      #pragma unroll
      for (int e = 0; e < 8; ++e) {
        qa[t][0][e] = q0[e]; qa[t][0][8 + e] = q1[e];
        qa[t][1][e] = q2[e]; qa[t][1][8 + e] = q3[e];
      }
    }

    for (int j = 0; j < NN; j += 32) {
      // next-tile prefetch -> global_prefetch_b8 (speculative, unguarded)
      __builtin_prefetch(Kh + (size_t)(j + 32 + l15) * DD, 0, 1);
      __builtin_prefetch(VTh + (size_t)(lane * 2) * NN + j + 32, 0, 1);

      // K^T B-fragments: rows of K are columns of K^T -> contiguous 32B loads
      const __bf16* krow0 = Kh + (size_t)(j + l15) * DD + lhi * 16;
      const __bf16* krow1 = Kh + (size_t)(j + 16 + l15) * DD + lhi * 16;
      const v16bf kb00 = *(const v16bf*)(krow0);
      const v16bf kb01 = *(const v16bf*)(krow0 + 32);
      const v16bf kb10 = *(const v16bf*)(krow1);
      const v16bf kb11 = *(const v16bf*)(krow1 + 32);

      // V B-fragments from V^T: contiguous 32B loads, shared by both i-tiles
      v16bf vb[4];
      #pragma unroll
      for (int dt = 0; dt < 4; ++dt)
        vb[dt] = *(const v16bf*)(VTh + (size_t)(dt * 16 + l15) * NN + j + lhi * 16);

      // S = Q' K^T ; relu ; convert ; stage bf16 for C->A transpose
      #pragma unroll
      for (int t = 0; t < 2; ++t) {
        v8f s0 = {}, s1 = {};
        s0 = wmma_bf16(qa[t][0], kb00, s0);
        s0 = wmma_bf16(qa[t][1], kb01, s0);
        s1 = wmma_bf16(qa[t][0], kb10, s1);
        s1 = wmma_bf16(qa[t][1], kb11, s1);
        __bf16* st = sstage[wave][t];
        #pragma unroll
        for (int r = 0; r < 8; ++r) {
          st[(r + 8 * lhi) * 32 + l15]      = (__bf16)fmaxf(s0[r], 0.0f);
          st[(r + 8 * lhi) * 32 + 16 + l15] = (__bf16)fmaxf(s1[r], 0.0f);
        }
      }
      // Per-wave buffer + in-order LDS pipe: only compiler ordering needed.
      __builtin_amdgcn_wave_barrier();
      v16bf pa[2];
      #pragma unroll
      for (int t = 0; t < 2; ++t) {
        const __bf16* st = sstage[wave][t];
        const v8bf p0 = *(const v8bf*)&st[l15 * 32 + lhi * 16];       // j-loc lo
        const v8bf p1 = *(const v8bf*)&st[l15 * 32 + lhi * 16 + 8];   // j-loc hi
        #pragma unroll
        for (int e = 0; e < 8; ++e) {
          pa[t][e] = p0[e]; pa[t][8 + e] = p1[e];
        }
      }
      __builtin_amdgcn_wave_barrier();   // protect staging reuse next iter

      // acc += P (16x32) * V (32x64); vb shared by both i-tiles
      #pragma unroll
      for (int dt = 0; dt < 4; ++dt) {
        acc[0][dt] = wmma_bf16(pa[0], vb[dt], acc[0][dt]);
        acc[1][dt] = wmma_bf16(pa[1], vb[dt], acc[1][dt]);
      }
    }
  }

  // Epilogue: residual add + f32 store (C layout: M = r + 8*lhi, N = l15).
  // 1/N was folded into Q', so acc is already correctly scaled.
  #pragma unroll
  for (int t = 0; t < 2; ++t) {
    #pragma unroll
    for (int dt = 0; dt < 4; ++dt) {
      #pragma unroll
      for (int r = 0; r < 8; ++r) {
        const size_t row = (size_t)rowBase + t * 16 + r + 8 * lhi;
        const size_t idx = ((size_t)b * NN + row) * DD + dt * 16 + l15;
        out[idx] = x[idx] + acc[t][dt][r];
      }
    }
  }
}

extern "C" void kernel_launch(void* const* d_in, const int* in_sizes, int n_in,
                              void* d_out, int out_size, void* d_ws, size_t ws_size,
                              hipStream_t stream) {
  const float* x  = (const float*)d_in[0];
  const float* WQ = (const float*)d_in[1];
  const float* WK = (const float*)d_in[2];
  const float* WV = (const float*)d_in[3];
  float* out = (float*)d_out;
  __bf16* ws = (__bf16*)d_ws;   // Q' | K | V^T, 3 * B*H*N*D bf16 = 12 MB

  qkv_project_kernel<<<BB * (NN / 64), 128, 0, stream>>>(x, WQ, WK, WV, ws);
  relu_attn_kernel<<<BB * (NN / 128), 128, 0, stream>>>(x, ws, out);
}
